// SelfAttention_63737314672780
// MI455X (gfx1250) — compile-verified, matrix operand order checked by reading, not run on hardware
//
#include <hip/hip_runtime.h>
#include <hip/hip_bf16.h>

typedef __attribute__((ext_vector_type(16))) __bf16 bf16x16;
typedef __attribute__((ext_vector_type(8)))  float  floatx8;
typedef __attribute__((ext_vector_type(4)))  unsigned uint32x4;
typedef __attribute__((ext_vector_type(8)))  unsigned uint32x8;

static constexpr int kS = 1024;   // sequence
static constexpr int kE = 1024;   // embed
static constexpr int kB = 4;      // batch
static constexpr int kH = 16;     // heads
static constexpr int kD = 64;     // head dim
static constexpr int kM = kB * kS;        // 4096 rows of x
static constexpr int kREL = 2 * kS - 1;   // 2047 relative positions

// ---------------- helpers ----------------

__device__ __forceinline__ __bf16 f2bf(float f) {
  unsigned u = __builtin_bit_cast(unsigned, f);
  u += 0x7FFFu + ((u >> 16) & 1u);                 // round-to-nearest-even
  unsigned short s = (unsigned short)(u >> 16);
  return __builtin_bit_cast(__bf16, s);
}

__device__ __forceinline__ floatx8 zero8() {
  floatx8 z;
#pragma unroll
  for (int i = 0; i < 8; ++i) z[i] = 0.f;
  return z;
}

// Scheduling fence: keep fragment loads and WMMA chains from interleaving
// (avoids per-WMMA s_wait_dscnt and WMMA->VALU WAR hazard v_nops).
__device__ __forceinline__ void sched_fence() {
#if __has_builtin(__builtin_amdgcn_sched_barrier)
  __builtin_amdgcn_sched_barrier(0);
#endif
}

// Async global -> LDS copy, 8 bytes per lane (CDNA5 GLOBAL_LOAD_ASYNC_TO_LDS_B64,
// tracked by ASYNCcnt). LDS destination is the low 32 bits of the generic
// pointer (generic LDS addr = {shared aperture, 32-bit LDS offset}).
__device__ __forceinline__ void async_ld_b64(void* lds_ptr, const void* gptr) {
  unsigned off = (unsigned)(size_t)lds_ptr;
  unsigned long long ga = (unsigned long long)(size_t)gptr;
  asm volatile("global_load_async_to_lds_b64 %0, %1, off"
               :
               : "v"(off), "v"(ga)
               : "memory");
}

__device__ __forceinline__ void wait_async() {
  asm volatile("s_wait_asynccnt 0x0" ::: "memory");
}

// Tensor Data Mover: DMA a 2D bf16 tile (tile_rows x 64 elems, global row
// stride 64 elems) into LDS with +2 DWORD padding per 32-DWORD row
// (=> LDS row stride 68 bf16). 2-group descriptor form (2D tensor,
// VADDR2/VADDR3 omitted). Tracked by TENSORcnt. Issue from one wave only.
__device__ __forceinline__ void tdm_load_2d(unsigned lds_off, const void* gaddr,
                                            unsigned tile_rows) {
  unsigned long long ga = (unsigned long long)(size_t)gaddr;
  uint32x4 g0;
  g0[0] = 1u;                                            // count=1, user D#
  g0[1] = lds_off;                                       // lds_addr (bytes)
  g0[2] = (unsigned)ga;                                  // global_addr[31:0]
  g0[3] = (unsigned)((ga >> 32) & 0x01FFFFFFu) | (2u << 30); // addr[56:32], type=2
  uint32x8 g1;
  g1[0] = (1u << 16)        // data_size = 2 bytes
        | (1u << 20)        // pad_enable
        | (4u << 22)        // pad_interval: 32 DWORDs (one 64-elem bf16 row)
        | (1u << 25);       // pad_amount: 2 DWORDs (4 bf16) -> ld 68
  g1[1] = (64u << 16);      // tensor_dim0 = 64 (bits 79:48 -> low half here)
  g1[2] = (1024u << 16);    // tensor_dim1 = 1024 (bits 111:80)
  g1[3] = (64u << 16);      // tile_dim0 = 64 (bits 127:112)
  g1[4] = tile_rows;        // tile_dim1 (bits 143:128); tile_dim2 = 0
  g1[5] = 64u;              // tensor_dim0_stride = 64 elems (bits 207:160)
  g1[6] = 0u;               // stride hi + tensor_dim1_stride lo
  g1[7] = 0u;
  asm volatile("tensor_load_to_lds %0, %1" :: "s"(g0), "s"(g1) : "memory");
}

__device__ __forceinline__ void wait_tensor() {
  asm volatile("s_wait_tensorcnt 0x0" ::: "memory");
}

union FragCast { bf16x16 v; uint2 u[4]; };

// Load a 16x32 bf16 fragment (A layout; also used for B via the mirrored
// role: lane&15 = output column, elements run along K contiguously).
// CDNA5 16-bit A layout: lanes 0-15 hold K=0..7 (elems0-7), K=16..23 (elems8-15);
// lanes 16-31 hold K=8..15 and K=24..31.
__device__ __forceinline__ bf16x16 load_frag(const __bf16* base, int row0, int ld, int k0) {
  const int lane = threadIdx.x & 31;
  const int hi = lane >> 4;
  const __bf16* p = base + (row0 + (lane & 15)) * ld + k0 + hi * 8;
  FragCast f;
  f.u[0] = *(const uint2*)(p);
  f.u[1] = *(const uint2*)(p + 4);
  f.u[2] = *(const uint2*)(p + 16);
  f.u[3] = *(const uint2*)(p + 20);
  return f.v;
}

__device__ __forceinline__ floatx8 wmma_bf16(bf16x16 a, bf16x16 b, floatx8 c) {
  return __builtin_amdgcn_wmma_f32_16x16x32_bf16(false, a, false, b, (short)0, c,
                                                 false, false);
}

// ---------------- fp32 -> bf16 conversion ----------------

__global__ __launch_bounds__(256) void cvt_bf16_kernel(const float* __restrict__ src,
                                                       __bf16* __restrict__ dst, int n) {
  int i = blockIdx.x * 256 + threadIdx.x;
  if (i < n) dst[i] = f2bf(src[i]);
}

// ---------------- T5 relative-position bias table: (H, 2047) ----------------

__global__ __launch_bounds__(256) void bias_table_kernel(const float* __restrict__ bias_emb,
                                                         float* __restrict__ table) {
  int r = blockIdx.x * 256 + threadIdx.x;
  if (r >= kREL) return;
  int rel = r - (kS - 1);                 // rel = k - q
  int rb = (rel > 0) ? 16 : 0;            // num_buckets/2
  int rp = (rel < 0) ? -rel : rel;
  int val;
  if (rp < 8) {
    val = rp;
  } else {
    // 8 + int( log(rp/8)/log(16) * 8 ), clamped to 15
    val = 8 + (int)(__logf((float)rp * 0.125f) * (8.0f / __logf(16.0f)));
    if (val > 15) val = 15;
  }
  int bucket = rb + val;
#pragma unroll
  for (int h = 0; h < kH; ++h)
    table[h * kREL + r] = bias_emb[bucket * kH + h];
}

// ---------------- expand bias table into (1,H,S,S) output ----------------

__global__ __launch_bounds__(256) void bias_out_kernel(const float* __restrict__ table,
                                                       float* __restrict__ outb) {
  unsigned i = blockIdx.x * 256u + threadIdx.x;   // < 16M
  int h = i >> 20;
  int q = (i >> 10) & 1023;
  int k = i & 1023;
  outb[i] = table[h * kREL + (k - q + (kS - 1))];
}

// ---------------- projection GEMMs: Q/K/V = x @ W.T (bf16, WMMA) -------------
// C[m,n] = sum_e x[m,e] * W[n,e]; output scattered to (B,H,S,D) bf16.
// Double-buffered LDS tiles staged with async-to-LDS copies.

__global__ __launch_bounds__(256) void gemm_qkv_kernel(const __bf16* __restrict__ xb,
                                                       const __bf16* __restrict__ wb,
                                                       __bf16* __restrict__ qb,
                                                       __bf16* __restrict__ kb,
                                                       __bf16* __restrict__ vb) {
  __shared__ __bf16 As[2][128 * 36];
  __shared__ __bf16 Bs[2][128 * 36];
  const int tid = threadIdx.x;
  const int wave = tid >> 5, lane = tid & 31;
  const int nlo = lane & 15, hi = lane >> 4;
  const int m0 = blockIdx.x * 128, n0 = blockIdx.y * 128;
  const int z = blockIdx.z;
  const __bf16* W = wb + (size_t)z * kE * kE;
  __bf16* dst = (z == 0) ? qb : (z == 1) ? kb : vb;

  auto stage = [&](int buf, int kk) {
#pragma unroll
    for (int it = 0; it < 4; ++it) {
      int idx = tid + it * 256;              // 1024 x b64 per matrix tile
      int row = idx >> 3, seg = idx & 7;
      async_ld_b64(&As[buf][row * 36 + seg * 4],
                   xb + (size_t)(m0 + row) * kE + kk + seg * 4);
      async_ld_b64(&Bs[buf][row * 36 + seg * 4],
                   W + (size_t)(n0 + row) * kE + kk + seg * 4);
    }
  };

  floatx8 acc[8];
#pragma unroll
  for (int i = 0; i < 8; ++i) acc[i] = zero8();

  stage(0, 0);
  for (int kk = 0; kk < kE; kk += 32) {
    const int cur = (kk >> 5) & 1;
    wait_async();
    __syncthreads();
    if (kk + 32 < kE) stage(cur ^ 1, kk + 32);

    // preload all fragments, then issue WMMAs back-to-back
    bf16x16 a = load_frag(As[cur], wave * 16, 36, 0);
    bf16x16 bfr[8];
#pragma unroll
    for (int nt = 0; nt < 8; ++nt) bfr[nt] = load_frag(Bs[cur], nt * 16, 36, 0);
    sched_fence();   // all DS loads above, all WMMAs below
#pragma unroll
    for (int nt = 0; nt < 8; ++nt) acc[nt] = wmma_bf16(a, bfr[nt], acc[nt]);
    sched_fence();
  }

#pragma unroll
  for (int nt = 0; nt < 8; ++nt) {
#pragma unroll
    for (int v = 0; v < 8; ++v) {
      int m = m0 + wave * 16 + v + 8 * hi;   // C layout: row = vgpr + 8*half
      int n = n0 + nt * 16 + nlo;            //            col = lane&15
      int b = m >> 10, s = m & 1023;
      int h = n >> 6, d = n & 63;
      dst[(size_t)((((b << 4) + h) << 10) + s) * kD + d] = f2bf(acc[nt][v]);
    }
  }
}

// ---------------- fused flash attention (bias + online softmax) --------------

struct FlashSmem {
  __bf16 k[2][64 * 68];     // double-buffered K tiles: [key][d] (ld 68), TDM-filled
  __bf16 vt[2][64 * 68];    // double-buffered V^T tiles: [d][key] (ld 68)
  union {
    __bf16 q[128 * 68];     // Q tile (TDM-filled); wave w reads rows w*16..w*16+15
    __bf16 p[8 * 16 * 68];  // per-wave P strips: same wave-local bytes, reused later
  } qp;
};

__global__ __launch_bounds__(256) void flash_attn_kernel(const __bf16* __restrict__ qb,
                                                         const __bf16* __restrict__ kbuf,
                                                         const __bf16* __restrict__ vbuf,
                                                         const float* __restrict__ table,
                                                         __bf16* __restrict__ ctx) {
  __shared__ FlashSmem sm;
  const int tid = threadIdx.x;
  const int wave = tid >> 5, lane = tid & 31;
  const int nlo = lane & 15, hi = lane >> 4;
  const int qblk = blockIdx.x & 7;
  const int h = (blockIdx.x >> 3) & 15;
  const int b = blockIdx.x >> 7;
  const size_t headBase = (size_t)((b << 4) + h) * kS * kD;
  const __bf16* Qg = qb + headBase + (size_t)qblk * 128 * kD;
  const __bf16* Kg = kbuf + headBase;
  const __bf16* Vg = vbuf + headBase;
  const float* tb = table + h * kREL;

  auto stageV = [&](int buf, int k0) {   // transpose in LDS: [d][key]
#pragma unroll
    for (int it = 0; it < 4; ++it) {
      int idx = tid + it * 256;
      int key = idx & 63, dseg = idx >> 6;
      union { uint2 u; __bf16 h4[4]; } t;
      t.u = *(const uint2*)(Vg + (size_t)(k0 + key) * kD + dseg * 4);
#pragma unroll
      for (int j = 0; j < 4; ++j) sm.vt[buf][(dseg * 4 + j) * 68 + key] = t.h4[j];
    }
  };

  // prologue: TDM-stage Q (128x64) and K tile 0 (64x64); manual V tile 0
  if (wave == 0) {
    tdm_load_2d((unsigned)(size_t)&sm.qp.q[0], Qg, 128u);
    tdm_load_2d((unsigned)(size_t)&sm.k[0][0], Kg, 64u);
  }
  stageV(0, 0);
  if (wave == 0) wait_tensor();
  __syncthreads();
  bf16x16 aQ0 = load_frag(sm.qp.q, wave * 16, 68, 0);
  bf16x16 aQ1 = load_frag(sm.qp.q, wave * 16, 68, 32);
  // wave w's P strip occupies the same wave-local bytes as its Q strip;
  // wave-local DS ops are in-order, so no extra barrier is needed.

  float mst[8], lst[8];
#pragma unroll
  for (int v = 0; v < 8; ++v) { mst[v] = -3.0e38f; lst[v] = 0.f; }
  floatx8 accO[4];
#pragma unroll
  for (int i = 0; i < 4; ++i) accO[i] = zero8();

  const int qrow0 = qblk * 128 + wave * 16 + 8 * hi;
  __bf16* Pw = sm.qp.p + wave * (16 * 68);

  for (int k0 = 0; k0 < kS; k0 += 64) {
    const int cur = (k0 >> 6) & 1;
    if (wave == 0) wait_tensor();   // K tile `cur` landed
    __syncthreads();                // V tile `cur` + everyone ready
    if (k0 + 64 < kS) {
      if (wave == 0)
        tdm_load_2d((unsigned)(size_t)&sm.k[cur ^ 1][0], Kg + (size_t)(k0 + 64) * kD, 64u);
      stageV(cur ^ 1, k0 + 64);
    }

    // scores: 16 q x 64 key, K-dim = D = 64; preload all K fragments first
    bf16x16 bk[8];
#pragma unroll
    for (int nt = 0; nt < 4; ++nt) {
      bk[nt]     = load_frag(sm.k[cur], nt * 16, 68, 0);
      bk[4 + nt] = load_frag(sm.k[cur], nt * 16, 68, 32);
    }
    sched_fence();   // all DS loads above, all WMMAs below
    floatx8 sc[4];
#pragma unroll
    for (int nt = 0; nt < 4; ++nt) sc[nt] = wmma_bf16(aQ0, bk[nt], zero8());
#pragma unroll
    for (int nt = 0; nt < 4; ++nt) sc[nt] = wmma_bf16(aQ1, bk[4 + nt], sc[nt]);
    sched_fence();

    // add T5 relative position bias from the (H,2047) table
#pragma unroll
    for (int nt = 0; nt < 4; ++nt) {
#pragma unroll
      for (int v = 0; v < 8; ++v) {
        int key = k0 + nt * 16 + nlo;
        int q = qrow0 + v;
        sc[nt][v] += tb[key - q + (kS - 1)];
      }
    }

    // online softmax, per q row (row v+8*hi lives in 16 lanes of a half-wave)
#pragma unroll
    for (int v = 0; v < 8; ++v) {
      float mx = sc[0][v];
#pragma unroll
      for (int nt = 1; nt < 4; ++nt) mx = fmaxf(mx, sc[nt][v]);
#pragma unroll
      for (int off = 8; off >= 1; off >>= 1) mx = fmaxf(mx, __shfl_xor(mx, off, 32));
      float mnew = fmaxf(mst[v], mx);
      float corr = __expf(mst[v] - mnew);
      mst[v] = mnew;
      float rs = 0.f;
#pragma unroll
      for (int nt = 0; nt < 4; ++nt) {
        float p = __expf(sc[nt][v] - mnew);
        sc[nt][v] = p;
        rs += p;
      }
#pragma unroll
      for (int off = 8; off >= 1; off >>= 1) rs += __shfl_xor(rs, off, 32);
      lst[v] = lst[v] * corr + rs;
#pragma unroll
      for (int dt = 0; dt < 4; ++dt) accO[dt][v] *= corr;
    }

    // C-layout -> A-layout relayout of P through wave-private LDS strip
#pragma unroll
    for (int nt = 0; nt < 4; ++nt) {
#pragma unroll
      for (int v = 0; v < 8; ++v)
        Pw[(v + 8 * hi) * 68 + nt * 16 + nlo] = f2bf(sc[nt][v]);
    }

    // accO += P (16x64) @ V (64 x 64d); preload fragments, then WMMA chain
    bf16x16 bv[8];
#pragma unroll
    for (int dt = 0; dt < 4; ++dt) {
      bv[dt]     = load_frag(sm.vt[cur], dt * 16, 68, 0);
      bv[4 + dt] = load_frag(sm.vt[cur], dt * 16, 68, 32);
    }
    bf16x16 aP0 = load_frag(Pw, 0, 68, 0);
    bf16x16 aP1 = load_frag(Pw, 0, 68, 32);
    sched_fence();   // all DS loads above, all WMMAs below
#pragma unroll
    for (int dt = 0; dt < 4; ++dt) accO[dt] = wmma_bf16(aP0, bv[dt], accO[dt]);
#pragma unroll
    for (int dt = 0; dt < 4; ++dt) accO[dt] = wmma_bf16(aP1, bv[4 + dt], accO[dt]);
    sched_fence();
  }

  // normalize and write context in (B,S,E) bf16 layout for the final GEMM
#pragma unroll
  for (int dt = 0; dt < 4; ++dt) {
#pragma unroll
    for (int v = 0; v < 8; ++v) {
      int q = qrow0 + v;
      int d = dt * 16 + nlo;
      float o = accO[dt][v] / lst[v];
      ctx[(size_t)((b << 10) + q) * kE + h * kD + d] = f2bf(o);
    }
  }
}

// ---------------- final GEMM: out = ctx @ Wo.T (fp32 output) ----------------

__global__ __launch_bounds__(256) void gemm_out_kernel(const __bf16* __restrict__ ab,
                                                       const __bf16* __restrict__ wo,
                                                       float* __restrict__ out) {
  __shared__ __bf16 As[2][128 * 36];
  __shared__ __bf16 Bs[2][128 * 36];
  const int tid = threadIdx.x;
  const int wave = tid >> 5, lane = tid & 31;
  const int nlo = lane & 15, hi = lane >> 4;
  const int m0 = blockIdx.x * 128, n0 = blockIdx.y * 128;

  auto stage = [&](int buf, int kk) {
#pragma unroll
    for (int it = 0; it < 4; ++it) {
      int idx = tid + it * 256;
      int row = idx >> 3, seg = idx & 7;
      async_ld_b64(&As[buf][row * 36 + seg * 4],
                   ab + (size_t)(m0 + row) * kE + kk + seg * 4);
      async_ld_b64(&Bs[buf][row * 36 + seg * 4],
                   wo + (size_t)(n0 + row) * kE + kk + seg * 4);
    }
  };

  floatx8 acc[8];
#pragma unroll
  for (int i = 0; i < 8; ++i) acc[i] = zero8();

  stage(0, 0);
  for (int kk = 0; kk < kE; kk += 32) {
    const int cur = (kk >> 5) & 1;
    wait_async();
    __syncthreads();
    if (kk + 32 < kE) stage(cur ^ 1, kk + 32);

    bf16x16 a = load_frag(As[cur], wave * 16, 36, 0);
    bf16x16 bfr[8];
#pragma unroll
    for (int nt = 0; nt < 8; ++nt) bfr[nt] = load_frag(Bs[cur], nt * 16, 36, 0);
    sched_fence();   // all DS loads above, all WMMAs below
#pragma unroll
    for (int nt = 0; nt < 8; ++nt) acc[nt] = wmma_bf16(a, bfr[nt], acc[nt]);
    sched_fence();
  }

#pragma unroll
  for (int nt = 0; nt < 8; ++nt) {
#pragma unroll
    for (int v = 0; v < 8; ++v) {
      int m = m0 + wave * 16 + v + 8 * hi;
      int n = n0 + nt * 16 + nlo;
      out[(size_t)m * kE + n] = acc[nt][v];
    }
  }
}

// ---------------- host launch ----------------

extern "C" void kernel_launch(void* const* d_in, const int* in_sizes, int n_in,
                              void* d_out, int out_size, void* d_ws, size_t ws_size,
                              hipStream_t stream) {
  (void)in_sizes; (void)n_in; (void)out_size; (void)ws_size;
  const float* x        = (const float*)d_in[0];
  const float* Wq       = (const float*)d_in[1];
  const float* Wk       = (const float*)d_in[2];
  const float* Wv       = (const float*)d_in[3];
  const float* Wo       = (const float*)d_in[4];
  const float* bias_emb = (const float*)d_in[5];

  float* out      = (float*)d_out;                    // (B,S,E) fp32, 4M elems
  float* out_bias = out + (size_t)kM * kE;            // (1,H,S,S) fp32, 16M elems

  char* ws = (char*)d_ws;
  __bf16* xb    = (__bf16*)(ws);                          //  8 MiB
  __bf16* wb    = (__bf16*)(ws + (size_t)(8)  * 1048576); //  8 MiB (Wq,Wk,Wv,Wo)
  __bf16* qb    = (__bf16*)(ws + (size_t)(16) * 1048576); //  8 MiB (B,H,S,D)
  __bf16* kb    = (__bf16*)(ws + (size_t)(24) * 1048576);
  __bf16* vb    = (__bf16*)(ws + (size_t)(32) * 1048576);
  __bf16* ctx   = (__bf16*)(ws + (size_t)(40) * 1048576); //  8 MiB (B,S,E)
  float*  table = (float*) (ws + (size_t)(48) * 1048576); // (H,2047) fp32

  const int nX = kM * kE;       // 4,194,304
  const int nW = kE * kE;       // 1,048,576

  cvt_bf16_kernel<<<nX / 256, 256, 0, stream>>>(x, xb, nX);
  cvt_bf16_kernel<<<nW / 256, 256, 0, stream>>>(Wq, wb + 0 * (size_t)nW, nW);
  cvt_bf16_kernel<<<nW / 256, 256, 0, stream>>>(Wk, wb + 1 * (size_t)nW, nW);
  cvt_bf16_kernel<<<nW / 256, 256, 0, stream>>>(Wv, wb + 2 * (size_t)nW, nW);
  cvt_bf16_kernel<<<nW / 256, 256, 0, stream>>>(Wo, wb + 3 * (size_t)nW, nW);

  bias_table_kernel<<<(kREL + 255) / 256, 256, 0, stream>>>(bias_emb, table);
  bias_out_kernel<<<(kH * kS * kS) / 256, 256, 0, stream>>>(table, out_bias);

  gemm_qkv_kernel<<<dim3(kM / 128, kE / 128, 3), 256, 0, stream>>>(xb, wb, qb, kb, vb);
  flash_attn_kernel<<<kB * kH * (kS / 128), 256, 0, stream>>>(qb, kb, vb, table, ctx);
  gemm_out_kernel<<<dim3(kM / 128, kE / 128), 256, 0, stream>>>(ctx, wb + 3 * (size_t)nW, out);
}
